// GraphSAGEModel_58798102282554
// MI455X (gfx1250) — matrix operand order repaired and checked by reading, not compiled, not run on hardware
//
#include <hip/hip_runtime.h>
#include <hip/hip_bf16.h>

typedef __bf16 bf16_t;
typedef __attribute__((ext_vector_type(16))) __bf16 v16bf;
typedef __attribute__((ext_vector_type(8)))  __bf16 v8bf;
typedef __attribute__((ext_vector_type(4)))  __bf16 v4bf;
typedef __attribute__((ext_vector_type(8)))  float  v8f;

#define DIM 128

// ---------------------------------------------------------------- zero fill
__global__ __launch_bounds__(256) void k_zero_f32(float* __restrict__ p, int n) {
    int i = blockIdx.x * 256 + threadIdx.x;
    if (i < n) p[i] = 0.0f;
}

// ------------------------------------------------- weights f32 -> bf16 copy
__global__ __launch_bounds__(256) void k_cvt_weights(const float* __restrict__ wl,
                                                     const float* __restrict__ wr,
                                                     bf16_t* __restrict__ wlb,
                                                     bf16_t* __restrict__ wrb, int n) {
    int i = blockIdx.x * 256 + threadIdx.x;
    if (i < n) {
        wlb[i] = (__bf16)wl[i];
        wrb[i] = (__bf16)wr[i];
    }
}

// ------------------------------------------ layer-1 scatter: x[src]->agg[dst]
// 32 lanes per edge, each lane handles 4 contiguous channels (float4 gather).
__global__ __launch_bounds__(256) void k_scatter_x(const float* __restrict__ x,
                                                   const long long* __restrict__ src,
                                                   const long long* __restrict__ dst,
                                                   float* __restrict__ agg,
                                                   float* __restrict__ deg, int n_edges) {
    int t = blockIdx.x * 256 + threadIdx.x;
    int e = t >> 5;
    if (e >= n_edges) return;
    int c = (t & 31) << 2;
    int s = (int)src[e];
    int d = (int)dst[e];
    const float4 v = *(const float4*)(x + (size_t)s * DIM + c);
    float* ap = agg + (size_t)d * DIM + c;
    atomicAdd(ap + 0, v.x);
    atomicAdd(ap + 1, v.y);
    atomicAdd(ap + 2, v.z);
    atomicAdd(ap + 3, v.w);
    if ((t & 31) == 0) atomicAdd(deg + d, 1.0f);
}

// --------------------------------------- layer-2 scatter: h_bf16[src]->agg[dst]
__global__ __launch_bounds__(256) void k_scatter_h(const bf16_t* __restrict__ h,
                                                   const long long* __restrict__ src,
                                                   const long long* __restrict__ dst,
                                                   float* __restrict__ agg, int n_edges) {
    int t = blockIdx.x * 256 + threadIdx.x;
    int e = t >> 5;
    if (e >= n_edges) return;
    int c = (t & 31) << 2;
    int s = (int)src[e];
    int d = (int)dst[e];
    v4bf v = *(const v4bf*)(h + (size_t)s * DIM + c);
    float* ap = agg + (size_t)d * DIM + c;
    atomicAdd(ap + 0, (float)v[0]);
    atomicAdd(ap + 1, (float)v[1]);
    atomicAdd(ap + 2, (float)v[2]);
    atomicAdd(ap + 3, (float)v[3]);
}

// ----------------------------------------------------------- layer-1 GEMM
// h = relu( (agg/deg) @ W1l^T + x @ W1r^T + b1 ), stored bf16.
// Wave-per-16x16-tile; 8 waves/WG -> 16 nodes x 128 features per block.
// A layout (16-bit A 16x32): lane = M row (lane%16); K chunks [h8*16, +8) and
// [h8*16+8, +8).  B layout (16-bit B 32x16): lane = N col; contiguous 16 K
// values starting at (lane/16)*16.  B[k][n] == W[n][k], so column n of B is
// row n of W -> straight 32B row loads from bf16 weights.
__global__ __launch_bounds__(256) void k_sage_gemm1(const float* __restrict__ x,
                                                    const float* __restrict__ agg,
                                                    const float* __restrict__ deg,
                                                    const bf16_t* __restrict__ w1l,
                                                    const bf16_t* __restrict__ w1r,
                                                    const float* __restrict__ b1,
                                                    bf16_t* __restrict__ h_out) {
    const int wave = threadIdx.x >> 5;   // 0..7 -> N tile
    const int lane = threadIdx.x & 31;
    const int kh   = lane >> 4;          // half-wave selector
    const int mr   = lane & 15;

    const int rowA = blockIdx.x * 16 + mr;   // node index (A matrix row)
    const int rowB = wave * 16 + mr;         // output feature (row of W)

    const float dinv = 1.0f / fmaxf(deg[rowA], 1.0f);

    const float*  xrow = x   + (size_t)rowA * DIM;
    const float*  arow = agg + (size_t)rowA * DIM;
    const bf16_t* wlp  = w1l + (size_t)rowB * DIM;
    const bf16_t* wrp  = w1r + (size_t)rowB * DIM;

    v8f acc = {};
#pragma unroll
    for (int ko = 0; ko < DIM; ko += 32) {
        const int k0 = ko + kh * 16;     // A chunk 0
        const int k1 = k0 + 8;           // A chunk 1
        v16bf aA, aX;
#pragma unroll
        for (int j = 0; j < 8; ++j) {
            aA[j]     = (__bf16)(arow[k0 + j] * dinv);
            aA[j + 8] = (__bf16)(arow[k1 + j] * dinv);
            aX[j]     = (__bf16)(xrow[k0 + j]);
            aX[j + 8] = (__bf16)(xrow[k1 + j]);
        }
        const v16bf bL = *(const v16bf*)(wlp + ko + kh * 16);
        const v16bf bR = *(const v16bf*)(wrp + ko + kh * 16);
        acc = __builtin_amdgcn_wmma_f32_16x16x32_bf16(false, aA, false, bL,
                                                      (short)0, acc, false, false);
        acc = __builtin_amdgcn_wmma_f32_16x16x32_bf16(false, aX, false, bR,
                                                      (short)0, acc, false, false);
    }

    // C/D layout: lane -> col = wave*16 + (lane%16); vgpr r -> row offset kh*8 + r
    const int   col  = wave * 16 + mr;
    const float bias = b1[col];
#pragma unroll
    for (int r = 0; r < 8; ++r) {
        const int rowO = blockIdx.x * 16 + kh * 8 + r;
        float v = acc[r] + bias;
        v = v > 0.0f ? v : 0.0f;
        h_out[(size_t)rowO * DIM + col] = (__bf16)v;
    }
}

// ------------------------------------------------------- layer-2 + sigmoid
// out[i] = sigmoid( (agg2[i]/deg[i]) . W2l + b2 + h[i] . W2r ), wave per node.
__global__ __launch_bounds__(256) void k_sage_out(const float* __restrict__ agg2,
                                                  const float* __restrict__ deg,
                                                  const bf16_t* __restrict__ h,
                                                  const float* __restrict__ w2l,
                                                  const float* __restrict__ b2,
                                                  const float* __restrict__ w2r,
                                                  float* __restrict__ out, int n) {
    const int wave = threadIdx.x >> 5;
    const int lane = threadIdx.x & 31;
    const int node = blockIdx.x * 8 + wave;
    if (node >= n) return;

    const float dinv = 1.0f / fmaxf(deg[node], 1.0f);
    const int   c    = lane * 4;
    const float4 a  = *(const float4*)(agg2 + (size_t)node * DIM + c);
    const v4bf   hh = *(const v4bf*)(h + (size_t)node * DIM + c);
    const float4 wl = *(const float4*)(w2l + c);
    const float4 wr = *(const float4*)(w2r + c);

    float s = dinv * (a.x * wl.x + a.y * wl.y + a.z * wl.z + a.w * wl.w)
            + (float)hh[0] * wr.x + (float)hh[1] * wr.y
            + (float)hh[2] * wr.z + (float)hh[3] * wr.w;

#pragma unroll
    for (int off = 16; off > 0; off >>= 1) s += __shfl_xor(s, off, 32);

    if (lane == 0) out[node] = 1.0f / (1.0f + __expf(-(s + b2[0])));
}

// ---------------------------------------------------------------- launcher
extern "C" void kernel_launch(void* const* d_in, const int* in_sizes, int n_in,
                              void* d_out, int out_size, void* d_ws, size_t ws_size,
                              hipStream_t stream) {
    const float*     x   = (const float*)d_in[0];
    const long long* ei  = (const long long*)d_in[1];
    const float*     W1l = (const float*)d_in[2];
    const float*     b1  = (const float*)d_in[3];
    const float*     W1r = (const float*)d_in[4];
    const float*     W2l = (const float*)d_in[5];
    const float*     b2  = (const float*)d_in[6];
    const float*     W2r = (const float*)d_in[7];
    float*           out = (float*)d_out;

    const int n_nodes = in_sizes[0] / DIM;       // 100000 (divisible by 16)
    const int n_edges = in_sizes[1] / 2;         // 1600000
    const long long* src = ei;
    const long long* dst = ei + n_edges;

    // workspace carve-up (aligned to 512B per region)
    auto align512 = [](size_t v) { return (v + 511) & ~(size_t)511; };
    char* p = (char*)d_ws;
    float*  agg   = (float*)p;  p += align512((size_t)n_nodes * DIM * sizeof(float));
    float*  deg   = (float*)p;  p += align512((size_t)n_nodes * sizeof(float));
    bf16_t* h_bf  = (bf16_t*)p; p += align512((size_t)n_nodes * DIM * sizeof(bf16_t));
    bf16_t* w1l_b = (bf16_t*)p; p += align512((size_t)DIM * DIM * sizeof(bf16_t));
    bf16_t* w1r_b = (bf16_t*)p;

    const int aggN      = n_nodes * DIM;
    const int zeroGrid  = (aggN + 255) / 256;
    const int degGrid   = (n_nodes + 255) / 256;
    const int edgeGrid  = (n_edges * 32 + 255) / 256;
    const int gemmGrid  = n_nodes / 16;
    const int outGrid   = (n_nodes + 7) / 8;
    const int wGrid     = (DIM * DIM + 255) / 256;

    // --- layer 1 ---
    k_zero_f32   <<<zeroGrid, 256, 0, stream>>>(agg, aggN);
    k_zero_f32   <<<degGrid,  256, 0, stream>>>(deg, n_nodes);
    k_cvt_weights<<<wGrid,    256, 0, stream>>>(W1l, W1r, w1l_b, w1r_b, DIM * DIM);
    k_scatter_x  <<<edgeGrid, 256, 0, stream>>>(x, src, dst, agg, deg, n_edges);
    k_sage_gemm1 <<<gemmGrid, 256, 0, stream>>>(x, agg, deg, w1l_b, w1r_b, b1, h_bf);

    // --- layer 2 (reuse agg buffer) ---
    k_zero_f32   <<<zeroGrid, 256, 0, stream>>>(agg, aggN);
    k_scatter_h  <<<edgeGrid, 256, 0, stream>>>(h_bf, src, dst, agg, n_edges);
    k_sage_out   <<<outGrid,  256, 0, stream>>>(agg, deg, h_bf, W2l, b2, W2r, out, n_nodes);
}
